// GatedDeltaNet_32847909879892
// MI455X (gfx1250) — compile-verified, matrix operand order checked by reading, not run on hardware
//
#include <hip/hip_runtime.h>
#include <hip/hip_bf16.h>

// ---------------------------------------------------------------------------
// GatedDeltaNet forward for MI455X (gfx1250, wave32, WMMA 16x16x32 f16,
// async global->LDS staging for the big GEMMs)
// ---------------------------------------------------------------------------

typedef __attribute__((ext_vector_type(16))) _Float16 v16h;
typedef __attribute__((ext_vector_type(8)))  _Float16 v8h;
typedef __attribute__((ext_vector_type(8)))  float    v8f;

#define DEV static __device__ __forceinline__

// Build a 16-half A/B fragment from two contiguous 8-half runs
// (k = klo..klo+7 and k = 16+klo..23+klo per the CDNA5 wave32 layout).
DEV v16h mkfrag(const _Float16* p0, const _Float16* p1) {
  v8h a = *(const v8h*)p0, b = *(const v8h*)p1;
  v16h r;
#pragma unroll
  for (int i = 0; i < 8; ++i) { r[i] = a[i]; r[i + 8] = b[i]; }
  return r;
}
DEV v16h mkfrag_s(const _Float16* p0, const _Float16* p1, float s) {
  v8h a = *(const v8h*)p0, b = *(const v8h*)p1;
  v16h r;
#pragma unroll
  for (int i = 0; i < 8; ++i) {
    r[i]     = (_Float16)((float)a[i] * s);
    r[i + 8] = (_Float16)((float)b[i] * s);
  }
  return r;
}
DEV v16h mkfrag_f32(const float* p0, const float* p1, float s) {
  v16h r;
#pragma unroll
  for (int i = 0; i < 8; ++i) {
    r[i]     = (_Float16)(p0[i] * s);
    r[i + 8] = (_Float16)(p1[i] * s);
  }
  return r;
}
DEV v8f wmma16(v16h a, v16h b, v8f c) {
  return __builtin_amdgcn_wmma_f32_16x16x32_f16(false, a, false, b, (short)0, c,
                                                false, false);
}

// CDNA5 async memory->LDS copy (16 bytes per lane, tracked with ASYNCcnt).
// lds_addr: 32-bit LDS byte address (low 32 bits of the generic pointer).
DEV void gdn_async_ld_b128(unsigned lds_addr, const void* gptr) {
  asm volatile("global_load_async_to_lds_b128 %0, %1, off"
               :
               : "v"(lds_addr), "v"(gptr)
               : "memory");
}
DEV void gdn_wait_async0() {
  asm volatile("s_wait_asynccnt 0x0" ::: "memory");
}

// ---------------------------------------------------------------------------
// elementwise converters
// ---------------------------------------------------------------------------
__global__ void gdn_cvt_f16(const float* __restrict__ s, _Float16* __restrict__ d,
                            size_t n) {
  for (size_t i = (size_t)blockIdx.x * blockDim.x + threadIdx.x; i < n;
       i += (size_t)gridDim.x * blockDim.x)
    d[i] = (_Float16)s[i];
}

// W[K][N] f32 -> Wt[N][K] f16
__global__ void gdn_transpose_f16(const float* __restrict__ W,
                                  _Float16* __restrict__ Wt, int K, int N) {
  __shared__ float tile[32][33];
  const int bn = blockIdx.x * 32, bk = blockIdx.y * 32;
  const int tx = threadIdx.x & 31, ty = threadIdx.x >> 5;
#pragma unroll
  for (int i = ty; i < 32; i += 8)
    tile[i][tx] = W[(size_t)(bk + i) * N + bn + tx];
  __syncthreads();
#pragma unroll
  for (int i = ty; i < 32; i += 8)
    Wt[(size_t)(bn + i) * K + bk + tx] = (_Float16)tile[tx][i];
}

// ---------------------------------------------------------------------------
// NT WMMA GEMM:  C[M][N] f32 = A[M][K] f16  *  Bt[N][K] f16 ^T
// block tile 128x128, 256 threads (8 waves), wave tile 32x64.
// Double-buffered LDS staged with GLOBAL_LOAD_ASYNC_TO_LDS_B128: the async
// copies for K-step k+1 are issued right after the barrier of step k, so
// global latency overlaps the 8 WMMAs of step k; the only stall is
// s_wait_asynccnt 0 + barrier at the top of the next step.
// ---------------------------------------------------------------------------
__launch_bounds__(256)
__global__ void gdn_gemm_nt(const _Float16* __restrict__ A,
                            const _Float16* __restrict__ Bt,
                            float* __restrict__ C, int M, int N, int K) {
  __shared__ _Float16 As[2][128 * 40];
  __shared__ _Float16 Bs[2][128 * 40];
  const int tid = threadIdx.x;
  const int lane = tid & 31, wave = tid >> 5;
  const int bm = blockIdx.x * 128, bn = blockIdx.y * 128;
  const int wm = (wave & 3) * 32;   // 2 m-subtiles
  const int wn = (wave >> 2) * 64;  // 4 n-subtiles
  const int rr = lane & 15;
  const int klo = (lane & 16) ? 8 : 0;

  // staging assignment: thread -> (row, 16-half chunk)
  const int sr = tid >> 1;
  const int sc = (tid & 1) * 16;
  const _Float16* gA = A + (size_t)(bm + sr) * K + sc;
  const _Float16* gB = Bt + (size_t)(bn + sr) * K + sc;
  unsigned lA[2], lB[2];
  lA[0] = (unsigned)(uintptr_t)&As[0][sr * 40 + sc];
  lA[1] = (unsigned)(uintptr_t)&As[1][sr * 40 + sc];
  lB[0] = (unsigned)(uintptr_t)&Bs[0][sr * 40 + sc];
  lB[1] = (unsigned)(uintptr_t)&Bs[1][sr * 40 + sc];

  v8f acc[2][4];
#pragma unroll
  for (int mi = 0; mi < 2; ++mi)
#pragma unroll
    for (int ni = 0; ni < 4; ++ni) acc[mi][ni] = (v8f){};

  // prologue: stage K-tile 0 into buffer 0
  gdn_async_ld_b128(lA[0],      gA);
  gdn_async_ld_b128(lA[0] + 16, gA + 8);
  gdn_async_ld_b128(lB[0],      gB);
  gdn_async_ld_b128(lB[0] + 16, gB + 8);

  int p = 0;
  for (int k0 = 0; k0 < K; k0 += 32) {
    gdn_wait_async0();   // our lane's async copies for buffer p landed in LDS
    __syncthreads();     // => everyone's copies landed; buffer p is ready
    if (k0 + 32 < K) {   // stage next K-tile into the other buffer
      const int q = p ^ 1;
      gdn_async_ld_b128(lA[q],      gA + k0 + 32);
      gdn_async_ld_b128(lA[q] + 16, gA + k0 + 40);
      gdn_async_ld_b128(lB[q],      gB + k0 + 32);
      gdn_async_ld_b128(lB[q] + 16, gB + k0 + 40);
    }
    if (k0 + 64 < K) {   // warm GL2 two tiles ahead (global_prefetch_b8)
      __builtin_prefetch(gA + k0 + 64, 0, 1);
      __builtin_prefetch(gB + k0 + 64, 0, 1);
    }

    const _Float16* Ab = As[p];
    const _Float16* Bb = Bs[p];
    v16h af[2], bf[4];
#pragma unroll
    for (int mi = 0; mi < 2; ++mi) {
      const _Float16* pp = Ab + (wm + mi * 16 + rr) * 40 + klo;
      af[mi] = mkfrag(pp, pp + 16);
    }
#pragma unroll
    for (int ni = 0; ni < 4; ++ni) {
      const _Float16* pp = Bb + (wn + ni * 16 + rr) * 40 + klo;
      bf[ni] = mkfrag(pp, pp + 16);
    }
#pragma unroll
    for (int mi = 0; mi < 2; ++mi)
#pragma unroll
      for (int ni = 0; ni < 4; ++ni)
        acc[mi][ni] = wmma16(af[mi], bf[ni], acc[mi][ni]);
    p ^= 1;
  }

  const int cn = lane & 15;
  const int mhi = (lane >> 4) * 8;
#pragma unroll
  for (int mi = 0; mi < 2; ++mi)
#pragma unroll
    for (int ni = 0; ni < 4; ++ni) {
      const int col = bn + wn + ni * 16 + cn;
#pragma unroll
      for (int v = 0; v < 8; ++v) {
        const int row = bm + wm + mi * 16 + mhi + v;
        C[(size_t)row * N + col] = acc[mi][ni][v];
      }
    }
}

// ---------------------------------------------------------------------------
// ba = x @ W_ba   (4096 x 2048 @ 2048 x 64, plain f32)
// ---------------------------------------------------------------------------
__global__ void gdn_ba(const float* __restrict__ x, const float* __restrict__ Wba,
                       float* __restrict__ ba) {
  const int idx = blockIdx.x * 256 + threadIdx.x;
  if (idx >= 4096 * 64) return;
  const int t = idx >> 6, j = idx & 63;
  const float* xr = x + (size_t)t * 2048;
  float s = 0.f;
  for (int k = 0; k < 2048; ++k) s += xr[k] * Wba[(size_t)k * 64 + j];
  ba[idx] = s;
}

// ---------------------------------------------------------------------------
// split qkvz [T][16*768] into mixed-flat (the reference's C-order reshape of
// [B,HK,T,512] to (B,8192,T)) and z [T][4096]
// ---------------------------------------------------------------------------
__global__ void gdn_split(const float* __restrict__ qkvz, float* __restrict__ mixed,
                          float* __restrict__ z) {
  for (size_t i = (size_t)blockIdx.x * 256 + threadIdx.x; i < 4096ULL * 12288ULL;
       i += (size_t)gridDim.x * 256) {
    const int t = (int)(i / 12288);
    const int rjs = (int)(i % 12288);
    const int hk = rjs / 768, j = rjs % 768;
    const float v = qkvz[i];
    if (j < 512)
      mixed[((size_t)hk * 4096 + t) * 512 + j] = v;       // flat [hk][t][512]
    else
      z[(size_t)t * 4096 + hk * 256 + (j - 512)] = v;     // z[t][hv*128+d]
  }
}

// ---------------------------------------------------------------------------
// depthwise causal conv (K=4) + SiLU over mixed viewed as [8192][4096]
// ---------------------------------------------------------------------------
__global__ void gdn_conv_silu(const float* __restrict__ mixed,
                              const float* __restrict__ cw, float* __restrict__ qb,
                              float* __restrict__ kb, float* __restrict__ vb) {
  const int t = blockIdx.x * 256 + threadIdx.x;
  const int c = blockIdx.y;
  if (t >= 4096) return;
  const float* m = mixed + (size_t)c * 4096;
  const float w0 = cw[c * 4 + 0], w1 = cw[c * 4 + 1];
  const float w2 = cw[c * 4 + 2], w3 = cw[c * 4 + 3];
  float a = m[t] * w3;
  if (t >= 1) a += m[t - 1] * w2;
  if (t >= 2) a += m[t - 2] * w1;
  if (t >= 3) a += m[t - 3] * w0;
  const float y = a / (1.f + __expf(-a));  // silu
  if (c < 2048)      qb[(size_t)t * 2048 + c] = y;
  else if (c < 4096) kb[(size_t)t * 2048 + (c - 2048)] = y;
  else               vb[(size_t)t * 4096 + (c - 4096)] = y;
}

// ---------------------------------------------------------------------------
// beta = sigmoid(b); g = -exp(A_log)*softplus(a + dt_bias)
// ---------------------------------------------------------------------------
__global__ void gdn_beta_g(const float* __restrict__ ba,
                           const float* __restrict__ A_log,
                           const float* __restrict__ dt_bias,
                           float* __restrict__ beta, float* __restrict__ g) {
  const int idx = blockIdx.x * 256 + threadIdx.x;
  if (idx >= 4096 * 32) return;
  const int t = idx >> 5, h = idx & 31;
  const int hk = h >> 1, r = h & 1;
  const float b = ba[(size_t)t * 64 + hk * 4 + r];
  const float a = ba[(size_t)t * 64 + hk * 4 + 2 + r];
  beta[idx] = 1.f / (1.f + __expf(-b));
  const float xx = a + dt_bias[h];
  const float sp = fmaxf(xx, 0.f) + log1pf(__expf(-fabsf(xx)));
  g[idx] = -__expf(A_log[h]) * sp;
}

// ---------------------------------------------------------------------------
// chunked gated delta rule: one block per v-head, sequential over 64 chunks.
// Dynamic LDS partition (bytes):
// ---------------------------------------------------------------------------
#define OFF_S    0         // float  S[128][128]      65536
#define OFF_ST   65536     // half   St[128][136]     34816  (St[n][k] = S^T)
#define OFF_QN   100352    // half   Qn[64][136]      17408
#define OFF_KN   117760    // half   Kn[64][136]      17408
#define OFF_XU   135168    // float  Xu[64][132]      33792  (vb -> u -> v_new)
#define OFF_XW   168960    // float  Xw[64][132]      33792  (kb*e^g -> w)
#define OFF_AA   202752    // float  Aa[64][68]       17408  (attn, decayed)
#define OFF_MM   220160    // float  Mm[64][65]       16640  (strict-tril M)
#define OFF_VNT  236800    // half   VnT[128][72]     18432  (v_new^T)
#define OFF_KDT  255232    // half   KdT[128][72]     18432  (kd^T)
#define OFF_GC   273664    // float  gc[64]
#define OFF_BET  273920    // float  bet[64]
#define OFF_MSC  274176    // float  misc[16]
#define GDN_DELTA_LDS 274240

__launch_bounds__(256)
__global__ void gdn_delta(const float* __restrict__ qb, const float* __restrict__ kb,
                          const float* __restrict__ vb,
                          const float* __restrict__ betab,
                          const float* __restrict__ gbuf,
                          float* __restrict__ obuf) {
  extern __shared__ char sm[];
  float*    S   = (float*)(sm + OFF_S);
  _Float16* St  = (_Float16*)(sm + OFF_ST);
  _Float16* Qn  = (_Float16*)(sm + OFF_QN);
  _Float16* Kn  = (_Float16*)(sm + OFF_KN);
  float*    Xu  = (float*)(sm + OFF_XU);
  float*    Xw  = (float*)(sm + OFF_XW);
  float*    Aa  = (float*)(sm + OFF_AA);
  float*    Mm  = (float*)(sm + OFF_MM);
  _Float16* VnT = (_Float16*)(sm + OFF_VNT);
  _Float16* KdT = (_Float16*)(sm + OFF_KDT);
  float*    gc  = (float*)(sm + OFF_GC);
  float*    bet = (float*)(sm + OFF_BET);
  float*    msc = (float*)(sm + OFF_MSC);

  const int tid = threadIdx.x;
  const int lane = tid & 31, wave = tid >> 5;
  const int h = blockIdx.x, hk = h >> 1;
  const int rr  = lane & 15;
  const int klo = (lane & 16) ? 8 : 0;
  const int cn  = lane & 15;
  const int mhi = (lane >> 4) * 8;

  for (int i = tid; i < 128 * 128; i += 256) S[i] = 0.f;
  __syncthreads();

  for (int nc = 0; nc < 64; ++nc) {
    const int t0 = nc * 64;
    // --- stage f16 transposed state copy ---
    for (int i = tid; i < 128 * 128; i += 256) {
      const int kk = i >> 7, nn = i & 127;
      St[nn * 136 + kk] = (_Float16)S[kk * 128 + nn];
    }
    if (tid < 64) {
      bet[tid] = betab[(size_t)(t0 + tid) * 32 + h];
      gc[tid]  = gbuf[(size_t)(t0 + tid) * 32 + h];
    }
    __syncthreads();
    if (tid == 0) {  // cumulative log-decay
      float s = 0.f;
      for (int i = 0; i < 64; ++i) { s += gc[i]; gc[i] = s; }
      msc[0] = s;
    }
    __syncthreads();
    const float gl = msc[0];

    // --- load q,k,v rows; l2norm q,k; Xu = v*beta ---
    if (tid < 64) {
      const int r = tid;
      const size_t tt = (size_t)(t0 + r);
      const float* qr = qb + tt * 2048 + hk * 128;
      const float* kr = kb + tt * 2048 + hk * 128;
      const float* vr = vb + tt * 4096 + h * 128;
      float sq = 0.f, sk = 0.f;
      for (int d = 0; d < 128; ++d) {
        const float a = qr[d]; sq += a * a;
        const float b = kr[d]; sk += b * b;
      }
      const float rq = rsqrtf(sq + 1e-6f) * 0.088388347648318447f;  // * Dk^-0.5
      const float rk = rsqrtf(sk + 1e-6f);
      const float bb = bet[r];
      for (int d = 0; d < 128; ++d) {
        Qn[r * 136 + d] = (_Float16)(qr[d] * rq);
        Kn[r * 136 + d] = (_Float16)(kr[d] * rk);
        Xu[r * 132 + d] = vr[d] * bb;
      }
    }
    __syncthreads();

    // --- WMMA: Mm = strict_tril((beta*Kn)Kn^T * decay), Aa = tril(Qn Kn^T * decay)
    for (int pass = 0; pass < 2; ++pass) {
      for (int tix = 0; tix < 2; ++tix) {
        const int tile = wave * 2 + tix;
        const int tm = (tile >> 2) * 16, tn = (tile & 3) * 16;
        v8f acc = (v8f){};
#pragma unroll
        for (int ks = 0; ks < 128; ks += 32) {
          const _Float16* pa = (pass == 0 ? Kn : Qn) + (tm + rr) * 136 + ks + klo;
          v16h af = (pass == 0) ? mkfrag_s(pa, pa + 16, bet[tm + rr])
                                : mkfrag(pa, pa + 16);
          const _Float16* pb = Kn + (tn + rr) * 136 + ks + klo;
          acc = wmma16(af, mkfrag(pb, pb + 16), acc);
        }
#pragma unroll
        for (int v = 0; v < 8; ++v) {
          const int i = tm + mhi + v, j = tn + cn;
          if (pass == 0)
            Mm[i * 65 + j] = (j < i) ? acc[v] * __expf(gc[i] - gc[j]) : 0.f;
          else
            Aa[i * 68 + j] = (j <= i) ? acc[v] * __expf(gc[i] - gc[j]) : 0.f;
        }
      }
    }
    // Xw = Kn*beta*exp(gcum)
    for (int i = tid; i < 64 * 128; i += 256) {
      const int r = i >> 7, d = i & 127;
      Xw[r * 132 + d] = (float)Kn[r * 136 + d] * bet[r] * __expf(gc[r]);
    }
    __syncthreads();

    // --- forward substitution: (I + Mm) X = RHS, unit lower triangular ---
    {
      float* Xc = (tid < 128) ? Xu : Xw;
      const int col = tid & 127;
      for (int i = 1; i < 64; ++i) {
        __syncthreads();
        float x = Xc[i * 132 + col];
        for (int j = 0; j < i; ++j) x -= Mm[i * 65 + j] * Xc[j * 132 + col];
        Xc[i * 132 + col] = x;
      }
    }
    __syncthreads();

    // --- v_new = u - w @ S  (A = -Xw, B = St, C = Xu) ---
    for (int i4 = 0; i4 < 4; ++i4) {
      const int tile = wave * 4 + i4;
      const int tm = (tile >> 3) * 16, tn = (tile & 7) * 16;
      v8f acc;
#pragma unroll
      for (int v = 0; v < 8; ++v) acc[v] = Xu[(tm + mhi + v) * 132 + tn + cn];
#pragma unroll
      for (int ks = 0; ks < 128; ks += 32) {
        const float* pa = Xw + (tm + rr) * 132 + ks + klo;
        v16h af = mkfrag_f32(pa, pa + 16, -1.f);
        const _Float16* pb = St + (tn + rr) * 136 + ks + klo;
        acc = wmma16(af, mkfrag(pb, pb + 16), acc);
      }
#pragma unroll
      for (int v = 0; v < 8; ++v) {
        const int r = tm + mhi + v, c2 = tn + cn;
        Xu[r * 132 + c2] = acc[v];
        VnT[c2 * 72 + r] = (_Float16)acc[v];
      }
    }
    __syncthreads();

    // KdT[d][c] = Kn[c][d]*exp(gl - gcum[c])
    for (int i = tid; i < 128 * 64; i += 256) {
      const int d = i >> 6, c = i & 63;
      KdT[d * 72 + c] = (_Float16)((float)Kn[c * 136 + d] * __expf(gl - gc[c]));
    }
    // --- o = qg @ S + attn @ v_new ---
    for (int i4 = 0; i4 < 4; ++i4) {
      const int tile = wave * 4 + i4;
      const int tm = (tile >> 3) * 16, tn = (tile & 7) * 16;
      v8f acc = (v8f){};
#pragma unroll
      for (int ks = 0; ks < 128; ks += 32) {
        const int row = tm + rr;
        const _Float16* pa = Qn + row * 136 + ks + klo;
        v16h af = mkfrag_s(pa, pa + 16, __expf(gc[row]));
        const _Float16* pb = St + (tn + rr) * 136 + ks + klo;
        acc = wmma16(af, mkfrag(pb, pb + 16), acc);
      }
#pragma unroll
      for (int ks = 0; ks < 64; ks += 32) {
        const float* pa = Aa + (tm + rr) * 68 + ks + klo;
        v16h af = mkfrag_f32(pa, pa + 16, 1.f);
        const _Float16* pb = VnT + (tn + rr) * 72 + ks + klo;
        acc = wmma16(af, mkfrag(pb, pb + 16), acc);
      }
#pragma unroll
      for (int v = 0; v < 8; ++v) {
        const int r = tm + mhi + v, c2 = tn + cn;
        obuf[(size_t)(t0 + r) * 4096 + h * 128 + c2] = acc[v];
      }
    }
    __syncthreads();

    // --- S = S*exp(gl) + kd^T @ v_new ---
    const float egl = __expf(gl);
    for (int i8 = 0; i8 < 8; ++i8) {
      const int tile = wave * 8 + i8;
      const int tm = (tile >> 3) * 16, tn = (tile & 7) * 16;
      v8f acc;
#pragma unroll
      for (int v = 0; v < 8; ++v) acc[v] = S[(tm + mhi + v) * 128 + tn + cn] * egl;
#pragma unroll
      for (int ks = 0; ks < 64; ks += 32) {
        const _Float16* pa = KdT + (tm + rr) * 72 + ks + klo;
        const _Float16* pb = VnT + (tn + rr) * 72 + ks + klo;
        acc = wmma16(mkfrag(pa, pa + 16), mkfrag(pb, pb + 16), acc);
      }
#pragma unroll
      for (int v = 0; v < 8; ++v) S[(tm + mhi + v) * 128 + tn + cn] = acc[v];
    }
    __syncthreads();
  }
}

// ---------------------------------------------------------------------------
// gated RMSNorm (per (t, head) over 128) + f16 convert for final GEMM
// ---------------------------------------------------------------------------
__global__ void gdn_gated_norm(const float* __restrict__ obuf,
                               const float* __restrict__ z,
                               const float* __restrict__ norm_w,
                               _Float16* __restrict__ og) {
  const int row = blockIdx.x * 256 + threadIdx.x;
  if (row >= 4096 * 32) return;
  const int t = row >> 5, h = row & 31;
  const float* o  = obuf + (size_t)t * 4096 + h * 128;
  const float* zz = z + (size_t)t * 4096 + h * 128;
  float ss = 0.f;
  for (int d = 0; d < 128; ++d) {
    const float zv = zz[d];
    const float v = o[d] * (zv / (1.f + __expf(-zv)));
    ss += v * v;
  }
  const float r = rsqrtf(ss * (1.f / 128.f) + 1e-6f);
  for (int d = 0; d < 128; ++d) {
    const float zv = zz[d];
    const float v = o[d] * (zv / (1.f + __expf(-zv))) * r * norm_w[d];
    og[(size_t)t * 4096 + h * 128 + d] = (_Float16)v;
  }
}

// ---------------------------------------------------------------------------
// host launcher
// ---------------------------------------------------------------------------
extern "C" void kernel_launch(void* const* d_in, const int* in_sizes, int n_in,
                              void* d_out, int out_size, void* d_ws, size_t ws_size,
                              hipStream_t stream) {
  (void)in_sizes; (void)n_in; (void)out_size; (void)ws_size;
  const float* x       = (const float*)d_in[0];
  const float* W_qkvz  = (const float*)d_in[3];
  const float* W_ba    = (const float*)d_in[4];
  const float* conv_w  = (const float*)d_in[5];
  const float* dt_bias = (const float*)d_in[6];
  const float* A_log   = (const float*)d_in[7];
  const float* norm_w  = (const float*)d_in[8];
  const float* W_out   = (const float*)d_in[9];
  float* out = (float*)d_out;

  char* w = (char*)d_ws;
  auto take = [&](size_t bytes) -> char* {
    char* p = w;
    w += (bytes + 255) & ~(size_t)255;
    return p;
  };
  _Float16* x_h  = (_Float16*)take(4096ULL * 2048 * 2);
  _Float16* WqT  = (_Float16*)take(12288ULL * 2048 * 2);
  _Float16* WoT  = (_Float16*)take(2048ULL * 4096 * 2);
  float* qkvz    = (float*)take(4096ULL * 12288 * 4);
  float* mixed   = (float*)take(8192ULL * 4096 * 4);
  float* zb      = (float*)take(4096ULL * 4096 * 4);
  float* bab     = (float*)take(4096ULL * 64 * 4);
  float* qb      = (float*)take(4096ULL * 2048 * 4);
  float* kb      = (float*)take(4096ULL * 2048 * 4);
  float* vb      = (float*)take(4096ULL * 4096 * 4);
  float* betab   = (float*)take(4096ULL * 32 * 4);
  float* gbuf    = (float*)take(4096ULL * 32 * 4);
  // qkvz is dead after gdn_split: reuse its 201MB region for o and og
  float* obuf    = qkvz;
  _Float16* og   = (_Float16*)((char*)qkvz + 4096ULL * 4096 * 4);

  gdn_cvt_f16<<<2048, 256, 0, stream>>>(x, x_h, 4096ULL * 2048);
  gdn_transpose_f16<<<dim3(384, 64), 256, 0, stream>>>(W_qkvz, WqT, 2048, 12288);
  gdn_transpose_f16<<<dim3(64, 128), 256, 0, stream>>>(W_out, WoT, 4096, 2048);

  gdn_gemm_nt<<<dim3(32, 96), 256, 0, stream>>>(x_h, WqT, qkvz, 4096, 12288, 2048);
  gdn_ba<<<1024, 256, 0, stream>>>(x, W_ba, bab);

  gdn_split<<<4096, 256, 0, stream>>>(qkvz, mixed, zb);
  gdn_conv_silu<<<dim3(16, 8192), 256, 0, stream>>>(mixed, conv_w, qb, kb, vb);
  gdn_beta_g<<<512, 256, 0, stream>>>(bab, A_log, dt_bias, betab, gbuf);

  gdn_delta<<<32, 256, GDN_DELTA_LDS, stream>>>(qb, kb, vb, betab, gbuf, obuf);

  gdn_gated_norm<<<512, 256, 0, stream>>>(obuf, zb, norm_w, og);
  gdn_gemm_nt<<<dim3(32, 16), 256, 0, stream>>>(og, WoT, out, 4096, 2048, 4096);
}
